// SpatialAttention_13434657702201
// MI455X (gfx1250) — matrix operand order repaired and checked by reading, not compiled, not run on hardware
//
#include <hip/hip_runtime.h>
#include <hip/hip_bf16.h>

// ---------------------------------------------------------------------------
// Problem constants (from the reference)
// ---------------------------------------------------------------------------
constexpr int BB = 2;        // batch
constexpr int HH = 64;
constexpr int WW = 64;
constexpr int CC = 128;
constexpr int NP = HH * WW;            // 4096 patches / pixels
constexpr int KD = 9 * CC;             // 1152 patch dim
constexpr long long NN = (long long)NP * NP;   // 16,777,216

typedef __attribute__((ext_vector_type(16))) __bf16 v16bf;
typedef __attribute__((ext_vector_type(8)))  __bf16 v8bf;
typedef __attribute__((ext_vector_type(8)))  float  v8f;
typedef unsigned int u32x4 __attribute__((ext_vector_type(4)));
typedef int          i32x4 __attribute__((ext_vector_type(4)));
typedef int          i32x8 __attribute__((ext_vector_type(8)));

#if defined(__gfx1250__) && __has_builtin(__builtin_amdgcn_tensor_load_to_lds)
#define HAVE_TDM 1
#endif

// fp32 -> bf16 (round to nearest even), stored as raw ushort
static __device__ __forceinline__ unsigned short f2bf(float f) {
  union { float f; unsigned u; } v; v.f = f;
  unsigned r = v.u + 0x7FFFu + ((v.u >> 16) & 1u);
  return (unsigned short)(r >> 16);
}
static __device__ __forceinline__ float bf2f(unsigned short h) {
  union { unsigned u; float f; } v; v.u = ((unsigned)h) << 16;
  return v.f;
}

#ifdef HAVE_TDM
// Issue one TDM 2-D tile load: 32 (K, bf16) x 128 (rows) from a row-major
// [rows, ldk] bf16 tensor into LDS at byte offset lds_off.  D# packing per
// CDNA5 ISA 8.3/8.4 (group0: count/lds/global/type; group1: dims+strides).
static __device__ __forceinline__ void tdm_tile_load(const unsigned short* g,
                                                     unsigned lds_off, int ldk) {
  const unsigned long long ga = (unsigned long long)(uintptr_t)g;
  u32x4 g0;
  g0[0] = 1u;                                              // count=1 (valid)
  g0[1] = lds_off;                                         // lds_addr (bytes)
  g0[2] = (unsigned)ga;                                    // global_addr[31:0]
  g0[3] = (unsigned)((ga >> 32) & 0x01FFFFFFu) | (2u << 30); // [56:32] | type=2
  i32x8 g1;
  g1[0] = (int)(1u << 16);          // workgroup_mask=0, data_size=1 (2 bytes)
  g1[1] = (int)(32u << 16);         // tensor_dim0 = 32   (bits 79:48 low half)
  g1[2] = (int)(128u << 16);        // tensor_dim0 hi=0 | tensor_dim1 = 128
  g1[3] = (int)(32u << 16);         // tensor_dim1 hi=0 | tile_dim0 = 32
  g1[4] = (int)128u;                // tile_dim1 = 128, tile_dim2 = 0
  g1[5] = ldk;                      // tensor_dim0_stride[31:0] = K
  g1[6] = 0;                        // stride0[47:32]=0, stride1 lo=0
  g1[7] = 0;                        // stride1 hi=0
  const i32x4 z4 = {0, 0, 0, 0};
#if __clang_major__ >= 23
  const i32x8 z8 = {0, 0, 0, 0, 0, 0, 0, 0};
  __builtin_amdgcn_tensor_load_to_lds(g0, g1, z4, z4, z8, 0);
#else
  __builtin_amdgcn_tensor_load_to_lds(g0, g1, z4, z4, 0);
#endif
}
#endif

// ---------------------------------------------------------------------------
// 1) Build patch matrices: P_bf16[b][q][1152], Pn_bf16 (L2-normalized),
//    mask validity mm[b][q].  One block (128 threads = channels) per patch.
// ---------------------------------------------------------------------------
__global__ __launch_bounds__(128)
void k_build_patches(const float* __restrict__ x, const float* __restrict__ mask,
                     unsigned short* __restrict__ P, unsigned short* __restrict__ Pn,
                     float* __restrict__ mm) {
  const int bq = blockIdx.x;
  const int b = bq / NP, q = bq % NP;
  const int h = q >> 6, w = q & 63;
  const int ch = threadIdx.x;

  float vals[9];
  float ss = 0.f;
#pragma unroll
  for (int ij = 0; ij < 9; ++ij) {
    const int i = ij / 3, j = ij % 3;
    const int hh = h + i - 1, ww = w + j - 1;
    float v = 0.f;
    if (hh >= 0 && hh < HH && ww >= 0 && ww < WW)
      v = x[(((size_t)b * NP) + (size_t)hh * WW + ww) * CC + ch];
    vals[ij] = v;
    ss += v * v;
  }

  __shared__ float red[128];
  red[ch] = ss;
  __syncthreads();
#pragma unroll
  for (int s = 64; s > 0; s >>= 1) {
    if (ch < s) red[ch] += red[ch + s];
    __syncthreads();
  }
  const float invn = 1.f / fmaxf(sqrtf(red[0]), 1e-4f);

  if (ch == 0) {
    float msum = 0.f;
#pragma unroll
    for (int ij = 0; ij < 9; ++ij) {
      const int hh = h + ij / 3 - 1, ww = w + ij % 3 - 1;
      if (hh >= 0 && hh < HH && ww >= 0 && ww < WW)
        msum += mask[(size_t)b * NP + (size_t)hh * WW + ww];
    }
    mm[bq] = (msum * (1.f / 9.f) == 1.0f) ? 1.f : 0.f;
  }

  const size_t base = (size_t)bq * KD;
#pragma unroll
  for (int ij = 0; ij < 9; ++ij) {
    P [base + (size_t)ij * CC + ch] = f2bf(vals[ij]);
    Pn[base + (size_t)ij * CC + ch] = f2bf(vals[ij] * invn);
  }
}

// ---------------------------------------------------------------------------
// 2) xT_bf16[b][c][n] = x[b][n][c]
// ---------------------------------------------------------------------------
__global__ __launch_bounds__(256)
void k_transpose_x(const float* __restrict__ x, unsigned short* __restrict__ xT) {
  const long long total = (long long)BB * CC * NP;
  for (long long e = (long long)blockIdx.x * blockDim.x + threadIdx.x; e < total;
       e += (long long)gridDim.x * blockDim.x) {
    const int b = (int)(e / ((long long)CC * NP));
    const long long r = e % ((long long)CC * NP);
    const int c = (int)(r / NP), n = (int)(r % NP);
    xT[e] = f2bf(x[((size_t)b * NP + n) * CC + c]);
  }
}

// ---------------------------------------------------------------------------
// 3) Double-buffered NT-GEMM with bf16 WMMA:  C[M][Nc] = A[M][K] * B[Nc][K]^T
//    128x128 block tile, 8 waves, each wave 32x64 (2x4 WMMA tiles), K step 32.
//    Next tile staged via TDM (tensor_load_to_lds) while WMMA runs; fallback
//    is register-staged global->LDS double buffering.
// ---------------------------------------------------------------------------
__global__ __launch_bounds__(256)
void k_gemm_nt(const unsigned short* __restrict__ A, long long sAb,
               const unsigned short* __restrict__ Bm, long long sBb,
               float* __restrict__ C, long long sCb,
               int K, int ldc) {
  __shared__ __align__(64) unsigned short As[2][128 * 32];
  __shared__ __align__(64) unsigned short Bs[2][128 * 32];

  const int b = blockIdx.z;
  const unsigned short* Ab = A  + (size_t)b * sAb + (size_t)blockIdx.x * 128 * K;
  const unsigned short* Bb = Bm + (size_t)b * sBb + (size_t)blockIdx.y * 128 * K;
  float* Cb = C + (size_t)b * sCb;

  const int tid  = threadIdx.x;
  const int lane = tid & 31;
  const int wid  = tid >> 5;
  const int waveM = wid >> 1;      // 0..3 -> rows  waveM*32
  const int waveN = wid & 1;       // 0..1 -> cols  waveN*64
  const int rr = lane & 15;
  const int hi = lane >> 4;

  v8f acc[8];
#pragma unroll
  for (int i = 0; i < 8; ++i)
#pragma unroll
    for (int j = 0; j < 8; ++j) acc[i][j] = 0.f;

  const int lrow = tid >> 2;             // 0..63
  const int lcol = (tid & 3) * 8;        // 0,8,16,24
  const int T = K / 32;

#ifdef HAVE_TDM
  // ---- TDM producer path: wave 0 drives the Tensor Data Mover -------------
  if (wid == 0) {
    tdm_tile_load(Ab, (unsigned)(uintptr_t)&As[0][0], K);
    tdm_tile_load(Bb, (unsigned)(uintptr_t)&Bs[0][0], K);
    __builtin_amdgcn_s_wait_tensorcnt(0);
  }
  __syncthreads();

  for (int t = 0; t < T; ++t) {
    const int cur = t & 1;
    if (t + 1 < T && wid == 0) {
      tdm_tile_load(Ab + (t + 1) * 32, (unsigned)(uintptr_t)&As[1 - cur][0], K);
      tdm_tile_load(Bb + (t + 1) * 32, (unsigned)(uintptr_t)&Bs[1 - cur][0], K);
    }
    if (t + 2 < T && wid == 1) {   // warm L2 for the tile after next
      __builtin_prefetch(Ab + (size_t)lrow * K + (t + 2) * 32, 0, 1);
      __builtin_prefetch(Bb + (size_t)lrow * K + (t + 2) * 32, 0, 1);
    }

    const unsigned short* as = As[cur];
    const unsigned short* bs = Bs[cur];
    v16bf afr[2];
#pragma unroll
    for (int mt = 0; mt < 2; ++mt) {
      const int row = waveM * 32 + mt * 16 + rr;
      const v8bf lo = *(const v8bf*)&as[row * 32 + hi * 16];
      const v8bf hh = *(const v8bf*)&as[row * 32 + hi * 16 + 8];
      afr[mt] = __builtin_shufflevector(lo, hh, 0, 1, 2, 3, 4, 5, 6, 7,
                                                8, 9, 10, 11, 12, 13, 14, 15);
    }
    v16bf bfr[4];
#pragma unroll
    for (int nt = 0; nt < 4; ++nt) {
      const int col = waveN * 64 + nt * 16 + rr;
      bfr[nt] = *(const v16bf*)&bs[col * 32 + hi * 16];
    }
#pragma unroll
    for (int mt = 0; mt < 2; ++mt)
#pragma unroll
      for (int nt = 0; nt < 4; ++nt)
        acc[mt * 4 + nt] = __builtin_amdgcn_wmma_f32_16x16x32_bf16(
            false, afr[mt], false, bfr[nt], (short)0, acc[mt * 4 + nt],
            false, false);

    if (t + 1 < T) {
      if (wid == 0) __builtin_amdgcn_s_wait_tensorcnt(0);
      __syncthreads();
    }
  }
#else
  // ---- Fallback: register-staged global->LDS double buffering -------------
  uint4 ra0, ra1, rb0, rb1;
  ra0 = *(const uint4*)(Ab + (size_t)lrow        * K + lcol);
  ra1 = *(const uint4*)(Ab + (size_t)(lrow + 64) * K + lcol);
  rb0 = *(const uint4*)(Bb + (size_t)lrow        * K + lcol);
  rb1 = *(const uint4*)(Bb + (size_t)(lrow + 64) * K + lcol);
  *(uint4*)&As[0][(size_t)tid * 8]        = ra0;
  *(uint4*)&As[0][2048 + (size_t)tid * 8] = ra1;
  *(uint4*)&Bs[0][(size_t)tid * 8]        = rb0;
  *(uint4*)&Bs[0][2048 + (size_t)tid * 8] = rb1;
  __syncthreads();

  for (int t = 0; t < T; ++t) {
    const int cur = t & 1;
    if (t + 1 < T) {
      const int k0 = (t + 1) * 32;
      ra0 = *(const uint4*)(Ab + (size_t)lrow        * K + k0 + lcol);
      ra1 = *(const uint4*)(Ab + (size_t)(lrow + 64) * K + k0 + lcol);
      rb0 = *(const uint4*)(Bb + (size_t)lrow        * K + k0 + lcol);
      rb1 = *(const uint4*)(Bb + (size_t)(lrow + 64) * K + k0 + lcol);
      if (t + 2 < T) {
        __builtin_prefetch(Ab + (size_t)lrow * K + k0 + 32, 0, 1);
        __builtin_prefetch(Bb + (size_t)lrow * K + k0 + 32, 0, 1);
      }
    }

    const unsigned short* as = As[cur];
    const unsigned short* bs = Bs[cur];
    v16bf afr[2];
#pragma unroll
    for (int mt = 0; mt < 2; ++mt) {
      const int row = waveM * 32 + mt * 16 + rr;
      const v8bf lo = *(const v8bf*)&as[row * 32 + hi * 16];
      const v8bf hh = *(const v8bf*)&as[row * 32 + hi * 16 + 8];
      afr[mt] = __builtin_shufflevector(lo, hh, 0, 1, 2, 3, 4, 5, 6, 7,
                                                8, 9, 10, 11, 12, 13, 14, 15);
    }
    v16bf bfr[4];
#pragma unroll
    for (int nt = 0; nt < 4; ++nt) {
      const int col = waveN * 64 + nt * 16 + rr;
      bfr[nt] = *(const v16bf*)&bs[col * 32 + hi * 16];
    }
#pragma unroll
    for (int mt = 0; mt < 2; ++mt)
#pragma unroll
      for (int nt = 0; nt < 4; ++nt)
        acc[mt * 4 + nt] = __builtin_amdgcn_wmma_f32_16x16x32_bf16(
            false, afr[mt], false, bfr[nt], (short)0, acc[mt * 4 + nt],
            false, false);

    if (t + 1 < T) {
      __syncthreads();   // everyone done reading buf (1-cur) from iter t-1
      const int nxt = 1 - cur;
      *(uint4*)&As[nxt][(size_t)tid * 8]        = ra0;
      *(uint4*)&As[nxt][2048 + (size_t)tid * 8] = ra1;
      *(uint4*)&Bs[nxt][(size_t)tid * 8]        = rb0;
      *(uint4*)&Bs[nxt][2048 + (size_t)tid * 8] = rb1;
      __syncthreads();
    }
  }
#endif

  // Epilogue: C/D layout — VGPR r: lanes 0-15 -> M=r, lanes 16-31 -> M=r+8
#pragma unroll
  for (int mt = 0; mt < 2; ++mt)
#pragma unroll
    for (int nt = 0; nt < 4; ++nt) {
      const int col = blockIdx.y * 128 + waveN * 64 + nt * 16 + rr;
#pragma unroll
      for (int r = 0; r < 8; ++r) {
        const int row = blockIdx.x * 128 + waveM * 32 + mt * 16 + r + (hi ? 8 : 0);
        Cb[(size_t)row * ldc + col] = acc[mt * 4 + nt][r];
      }
    }
}

// ---------------------------------------------------------------------------
// 4) fuse_diag on flattened [N][N]
// ---------------------------------------------------------------------------
__global__ __launch_bounds__(256)
void k_fuse1(const float* __restrict__ in, float* __restrict__ out) {
  const long long total = (long long)BB * NN;
  for (long long e = (long long)blockIdx.x * blockDim.x + threadIdx.x; e < total;
       e += (long long)gridDim.x * blockDim.x) {
    const int b = (int)(e / NN);
    const long long r = e % NN;
    const int a = (int)(r / NP), j = (int)(r % NP);
    const float* base = in + (size_t)b * NN;
    float s = base[(size_t)a * NP + j];
    if (a > 0 && j > 0)             s += base[(size_t)(a - 1) * NP + (j - 1)];
    if (a < NP - 1 && j < NP - 1)   s += base[(size_t)(a + 1) * NP + (j + 1)];
    out[e] = s;
  }
}

// out[p][q] = sum_{d=-1..1} in[T(T(p)+d)][T(T(q)+d)],  T(n) = (n%64)*64 + n/64
__global__ __launch_bounds__(256)
void k_fuse2t(const float* __restrict__ in, float* __restrict__ out) {
  const long long total = (long long)BB * NN;
  for (long long e = (long long)blockIdx.x * blockDim.x + threadIdx.x; e < total;
       e += (long long)gridDim.x * blockDim.x) {
    const int b = (int)(e / NN);
    const long long r = e % NN;
    const int p = (int)(r / NP), q = (int)(r % NP);
    const int tp = (p & 63) * 64 + (p >> 6);
    const int tq = (q & 63) * 64 + (q >> 6);
    const float* base = in + (size_t)b * NN;
    float s = 0.f;
#pragma unroll
    for (int d = -1; d <= 1; ++d) {
      const int ap = tp + d, aq = tq + d;
      if (ap >= 0 && ap < NP && aq >= 0 && aq < NP) {
        const int ip = (ap & 63) * 64 + (ap >> 6);
        const int iq = (aq & 63) * 64 + (aq >> 6);
        s += base[(size_t)ip * NP + iq];
      }
    }
    out[e] = s;
  }
}

// ---------------------------------------------------------------------------
// 5) Masked softmax over q (scale 10):  v=s*mm ; p=softmax(10*v) ; p*=mm
// ---------------------------------------------------------------------------
__global__ __launch_bounds__(256)
void k_softmax(const float* __restrict__ S, const float* __restrict__ mm,
               unsigned short* __restrict__ ybf) {
  const int bq = blockIdx.x;           // b*NP + p
  const int b = bq / NP;
  const int tid = threadIdx.x, lane = tid & 31, wid = tid >> 5;
  const float* row = S + (size_t)bq * NP;
  const float* mrow = mm + (size_t)b * NP;

  float vals[16];
  float tmax = -3.4e38f;
#pragma unroll
  for (int i = 0; i < 16; ++i) {
    const int q = tid + i * 256;
    const float v = row[q] * mrow[q];
    vals[i] = v;
    tmax = fmaxf(tmax, v);
  }
#pragma unroll
  for (int o = 16; o >= 1; o >>= 1) tmax = fmaxf(tmax, __shfl_xor(tmax, o));

  __shared__ float sred[8];
  if (lane == 0) sred[wid] = tmax;
  __syncthreads();
  float rmax = sred[0];
#pragma unroll
  for (int i = 1; i < 8; ++i) rmax = fmaxf(rmax, sred[i]);
  const float M = 10.f * rmax;

  float tsum = 0.f;
#pragma unroll
  for (int i = 0; i < 16; ++i) {
    vals[i] = __expf(10.f * vals[i] - M);
    tsum += vals[i];
  }
#pragma unroll
  for (int o = 16; o >= 1; o >>= 1) tsum += __shfl_xor(tsum, o);
  __syncthreads();
  if (lane == 0) sred[wid] = tsum;
  __syncthreads();
  float rsum = 0.f;
#pragma unroll
  for (int i = 0; i < 8; ++i) rsum += sred[i];
  const float inv = 1.f / rsum;

  unsigned short* orow = ybf + (size_t)bq * NP;
#pragma unroll
  for (int i = 0; i < 16; ++i) {
    const int q = tid + i * 256;
    orow[q] = f2bf(vals[i] * inv * mrow[q]);
  }
}

// ---------------------------------------------------------------------------
// 6) Z[p][r] = sum_{sy,sx in {-1,0,1}} y[p+s][r+s] -> transposed conv as GEMM
// ---------------------------------------------------------------------------
__global__ __launch_bounds__(256)
void k_zfuse(const unsigned short* __restrict__ ybf, unsigned short* __restrict__ zbf) {
  const long long total = (long long)BB * NN;
  for (long long e = (long long)blockIdx.x * blockDim.x + threadIdx.x; e < total;
       e += (long long)gridDim.x * blockDim.x) {
    const int b = (int)(e / NN);
    const long long r0 = e % NN;
    const int p = (int)(r0 / NP), r = (int)(r0 % NP);
    const int ph = p >> 6, pw = p & 63, rh = r >> 6, rw = r & 63;
    const unsigned short* base = ybf + (size_t)b * NN;
    float acc = 0.f;
#pragma unroll
    for (int sy = -1; sy <= 1; ++sy)
#pragma unroll
      for (int sx = -1; sx <= 1; ++sx) {
        const int ah = ph + sy, aw = pw + sx, bh = rh + sy, bw = rw + sx;
        if (ah >= 0 && ah < HH && aw >= 0 && aw < WW &&
            bh >= 0 && bh < HH && bw >= 0 && bw < WW)
          acc += bf2f(base[(size_t)(ah * WW + aw) * NP + (bh * WW + bw)]);
      }
    zbf[e] = f2bf(acc);
  }
}

// ---------------------------------------------------------------------------
// 7) Epilogue prep: weight transposes, feature patches, gating
// ---------------------------------------------------------------------------
__global__ __launch_bounds__(256)
void k_wT(const float* __restrict__ src, unsigned short* __restrict__ dst, int KIN) {
  const long long total = (long long)128 * KIN;
  for (long long e = (long long)blockIdx.x * blockDim.x + threadIdx.x; e < total;
       e += (long long)gridDim.x * blockDim.x) {
    const int c = (int)(e / KIN), k = (int)(e % KIN);
    dst[e] = f2bf(src[(size_t)k * 128 + c]);
  }
}

__global__ __launch_bounds__(256)
void k_featpatch(const float* __restrict__ attn, const float* __restrict__ x,
                 unsigned short* __restrict__ fp) {
  const long long total = (long long)BB * NP * 2304;
  for (long long e = (long long)blockIdx.x * blockDim.x + threadIdx.x; e < total;
       e += (long long)gridDim.x * blockDim.x) {
    const int bp = (int)(e / 2304), k = (int)(e % 2304);
    const int ij = k / 256, ch = k % 256;
    const int b = bp / NP, p = bp % NP;
    const int hh = (p >> 6) + ij / 3 - 1, ww = (p & 63) + ij % 3 - 1;
    float v = 0.f;
    if (hh >= 0 && hh < HH && ww >= 0 && ww < WW) {
      const size_t sp = (size_t)b * NP + (size_t)hh * WW + ww;
      v = (ch < 128) ? attn[sp * CC + ch] : x[sp * CC + (ch - 128)];
    }
    fp[e] = f2bf(v);
  }
}

__global__ __launch_bounds__(256)
void k_featcenter(const float* __restrict__ attn, const float* __restrict__ x,
                  unsigned short* __restrict__ fc) {
  const long long total = (long long)BB * NP * 256;
  for (long long e = (long long)blockIdx.x * blockDim.x + threadIdx.x; e < total;
       e += (long long)gridDim.x * blockDim.x) {
    const long long bp = e / 256;
    const int ch = (int)(e % 256);
    fc[e] = f2bf((ch < 128) ? attn[bp * CC + ch] : x[bp * CC + (ch - 128)]);
  }
}

__global__ __launch_bounds__(256)
void k_gate(const float* __restrict__ oc, const float* __restrict__ o1,
            const float* __restrict__ o2,
            const float* __restrict__ bc, const float* __restrict__ b1,
            const float* __restrict__ b2, float* __restrict__ out) {
  const long long total = (long long)BB * NP * CC;
  for (long long e = (long long)blockIdx.x * blockDim.x + threadIdx.x; e < total;
       e += (long long)gridDim.x * blockDim.x) {
    const int c = (int)(e % CC);
    const float g1 = 1.f / (1.f + __expf(-(o1[e] + b1[c])));
    const float g2 = 1.f / (1.f + __expf(-(o2[e] + b2[c])));
    out[e] = 0.5f * (g1 + g2) * (oc[e] + bc[c]);
  }
}

// ---------------------------------------------------------------------------
// Launch
// ---------------------------------------------------------------------------
static constexpr size_t alup(size_t x) { return (x + 255) & ~(size_t)255; }

extern "C" void kernel_launch(void* const* d_in, const int* in_sizes, int n_in,
                              void* d_out, int out_size, void* d_ws, size_t ws_size,
                              hipStream_t stream) {
  (void)in_sizes; (void)n_in; (void)out_size; (void)ws_size;
  const float* x    = (const float*)d_in[0];
  const float* mask = (const float*)d_in[1];
  const float* Wc   = (const float*)d_in[2];
  const float* bc   = (const float*)d_in[3];
  const float* W1   = (const float*)d_in[4];
  const float* b1   = (const float*)d_in[5];
  const float* W2   = (const float*)d_in[6];
  const float* b2   = (const float*)d_in[7];
  float* out = (float*)d_out;
  char* ws = (char*)d_ws;

  size_t off = 0;
  auto take = [&](size_t bytes) { size_t o = off; off = alup(off + bytes); return o; };
  const size_t szPat = (size_t)BB * NP * KD * 2;
  unsigned short* Pbf  = (unsigned short*)(ws + take(szPat));
  unsigned short* Pnbf = (unsigned short*)(ws + take(szPat));
  float* S0  = (float*)(ws + take((size_t)BB * NN * 4));
  float* S1  = (float*)(ws + take((size_t)BB * NN * 4));
  unsigned short* ybf = (unsigned short*)(ws + take((size_t)BB * NN * 2));
  unsigned short* zbf = (unsigned short*)(ws + take((size_t)BB * NN * 2));
  float* mm  = (float*)(ws + take((size_t)BB * NP * 4));
  unsigned short* xT  = (unsigned short*)(ws + take((size_t)BB * CC * NP * 2));
  float* attn = (float*)(ws + take((size_t)BB * NP * CC * 4));
  unsigned short* fpch = (unsigned short*)(ws + take((size_t)BB * NP * 2304 * 2));
  unsigned short* fctr = (unsigned short*)(ws + take((size_t)BB * NP * 256 * 2));
  unsigned short* WcT = (unsigned short*)(ws + take((size_t)128 * 2304 * 2));
  unsigned short* W2T = (unsigned short*)(ws + take((size_t)128 * 2304 * 2));
  unsigned short* W1T = (unsigned short*)(ws + take((size_t)128 * 256 * 2));
  float* oc = (float*)(ws + take((size_t)BB * NP * CC * 4));
  float* o1 = (float*)(ws + take((size_t)BB * NP * CC * 4));
  float* o2 = (float*)(ws + take((size_t)BB * NP * CC * 4));

  const long long elemsNN = (long long)BB * NN;
  const int gsNN = (int)((elemsNN + 255) / 256);

  k_build_patches<<<BB * NP, 128, 0, stream>>>(x, mask, Pbf, Pnbf, mm);
  k_transpose_x<<<(BB * CC * NP + 255) / 256, 256, 0, stream>>>(x, xT);

  // GEMM1: score = P * Pn^T   [per batch: 4096 x 4096 x 1152]
  k_gemm_nt<<<dim3(NP / 128, NP / 128, BB), 256, 0, stream>>>(
      Pbf, (long long)NP * KD, Pnbf, (long long)NP * KD,
      S0, NN, KD, NP);

  k_fuse1 <<<gsNN, 256, 0, stream>>>(S0, S1);
  k_fuse2t<<<gsNN, 256, 0, stream>>>(S1, S0);
  k_softmax<<<BB * NP, 256, 0, stream>>>(S0, mm, ybf);
  k_zfuse<<<gsNN, 256, 0, stream>>>(ybf, zbf);

  // GEMM2: attn = Z * x  [4096 x 128 x 4096]
  k_gemm_nt<<<dim3(NP / 128, 1, BB), 256, 0, stream>>>(
      zbf, NN, xT, (long long)CC * NP,
      attn, (long long)NP * CC, NP, CC);

  k_wT<<<(128 * 2304 + 255) / 256, 256, 0, stream>>>(Wc, WcT, 2304);
  k_wT<<<(128 * 2304 + 255) / 256, 256, 0, stream>>>(W2, W2T, 2304);
  k_wT<<<(128 * 256 + 255) / 256, 256, 0, stream>>>(W1, W1T, 256);
  k_featpatch <<<(int)(((long long)BB * NP * 2304 + 255) / 256), 256, 0, stream>>>(attn, x, fpch);
  k_featcenter<<<(int)(((long long)BB * NP * 256 + 255) / 256), 256, 0, stream>>>(attn, x, fctr);

  // 8192 x 128 x {2304, 2304, 256}
  k_gemm_nt<<<dim3(BB * NP / 128, 1, 1), 256, 0, stream>>>(
      fpch, 0, WcT, 0, oc, 0, 2304, CC);
  k_gemm_nt<<<dim3(BB * NP / 128, 1, 1), 256, 0, stream>>>(
      fpch, 0, W2T, 0, o2, 0, 2304, CC);
  k_gemm_nt<<<dim3(BB * NP / 128, 1, 1), 256, 0, stream>>>(
      fctr, 0, W1T, 0, o1, 0, 256, CC);

  k_gate<<<(int)(((long long)BB * NP * CC + 255) / 256), 256, 0, stream>>>(
      oc, o1, o2, bc, b1, b2, out);
}
